// CrossAttention_78305843740743
// MI455X (gfx1250) — matrix operand order repaired and checked by reading, not compiled, not run on hardware
//
#include <hip/hip_runtime.h>
#include <math.h>

#define D_ATT 128
#define D_IN  512
#define NQ    256
#define NK    2048
#define BSZ   2

typedef __attribute__((ext_vector_type(16))) _Float16 v16h;
typedef __attribute__((ext_vector_type(8)))  _Float16 v8h;
typedef __attribute__((ext_vector_type(8)))  float    v8f;

__device__ __forceinline__ float fast_tanh(float x) {
#if __has_builtin(__builtin_amdgcn_tanhf)
    return __builtin_amdgcn_tanhf(x);            // native V_TANH_F32 (TRANS32)
#elif __has_builtin(__builtin_amdgcn_tanh_f32)
    return __builtin_amdgcn_tanh_f32(x);
#else
    return tanhf(x);
#endif
}

// ---------------------------------------------------------------------------
// Pack W [512,128] f32 (row-major) into WMMA B-fragment order, f16:
//   frag = (nt*16 + ks)*2 + half   (nt: N-tile 0..7, ks: K-step 0..15, half: 0..1)
//   Wp[frag*256 + l15*16 + j] = W[(ks*32 + (j&8)*2 + half*8 + (j&7))*128 + nt*16 + l15]
// so a lane's 16 B-fragment halves are 32 contiguous bytes.
// Launch: 256 blocks x 256 threads = 65536 elements.
// ---------------------------------------------------------------------------
__global__ __launch_bounds__(256) void packw_kernel(
    const float* __restrict__ W, _Float16* __restrict__ Wp)
{
    const int t    = blockIdx.x * 256 + threadIdx.x;   // 0 .. 65535
    const int j    = t & 15;
    const int l15  = (t >> 4) & 15;
    const int half = (t >> 8) & 1;
    const int ks   = (t >> 9) & 15;
    const int nt   = t >> 13;
    const int k = ks * 32 + (j & 8) * 2 + half * 8 + (j & 7);
    const int n = nt * 16 + l15;
    Wp[t] = (_Float16)W[(size_t)k * D_ATT + n];
}

// ---------------------------------------------------------------------------
// Projection: Y[M,128] = X[M,512] @ W[512,128] + bias.
// One wave per 16x16 tile; K swept 32/step with v_wmma_f32_16x16x32_f16.
// A: f32 loaded 128b and cvt_pk'd to f16; B: pre-packed f16, contiguous.
// OUT_HALF=0 -> f32 output (tanh path), OUT_HALF=1 -> f16 output (bilinear).
// Grid must be exactly M/16 blocks of 256 threads (8 waves = 8 N-tiles).
// ---------------------------------------------------------------------------
template <int OUT_HALF>
__global__ __launch_bounds__(256) void proj_kernel(
    const float* __restrict__ X, const _Float16* __restrict__ Wp,
    const float* __restrict__ bias, void* __restrict__ outv)
{
    const int lane = threadIdx.x & 31;
    const int wave = blockIdx.x * 8 + (threadIdx.x >> 5);
    const int mt   = wave >> 3;        // M tile
    const int nt   = wave & 7;         // N tile (128/16 = 8)
    const int half = lane >> 4;
    const int l15  = lane & 15;

    const int arow = mt * 16 + l15;    // A-matrix row for this lane
    const int n    = nt * 16 + l15;    // output column for this lane

    v8f acc = {};
    for (int ks = 0; ks < 16; ++ks) {
        const int kk = ks * 32;
        // A fragment: row `arow`, K chunks [kk+half*8, +8) and [kk+16+half*8, +8)
        const float* ap = X + (size_t)arow * D_IN + kk + half * 8;
        float4 f0 = *(const float4*)(ap);
        float4 f1 = *(const float4*)(ap + 4);
        float4 f2 = *(const float4*)(ap + 16);
        float4 f3 = *(const float4*)(ap + 20);
        v16h a;
        a[0]=(_Float16)f0.x;  a[1]=(_Float16)f0.y;  a[2]=(_Float16)f0.z;  a[3]=(_Float16)f0.w;
        a[4]=(_Float16)f1.x;  a[5]=(_Float16)f1.y;  a[6]=(_Float16)f1.z;  a[7]=(_Float16)f1.w;
        a[8]=(_Float16)f2.x;  a[9]=(_Float16)f2.y;  a[10]=(_Float16)f2.z; a[11]=(_Float16)f2.w;
        a[12]=(_Float16)f3.x; a[13]=(_Float16)f3.y; a[14]=(_Float16)f3.z; a[15]=(_Float16)f3.w;

        // B fragment: pre-packed, 32 contiguous bytes per lane
        const _Float16* bp = Wp + (size_t)(((nt * 16 + ks) * 2 + half) * 16 + l15) * 16;
        v8h b0 = *(const v8h*)(bp);
        v8h b1 = *(const v8h*)(bp + 8);
        v16h b;
        #pragma unroll
        for (int j = 0; j < 8; ++j) { b[j] = b0[j]; b[j + 8] = b1[j]; }

        acc = __builtin_amdgcn_wmma_f32_16x16x32_f16(
            false, a, false, b, (short)0, acc, false, false);
    }

    const float bn = bias[n];
    if constexpr (OUT_HALF) {
        _Float16* out = (_Float16*)outv;
        #pragma unroll
        for (int r = 0; r < 8; ++r) {
            const int orow = mt * 16 + half * 8 + r;
            out[(size_t)orow * D_ATT + n] = (_Float16)(acc[r] + bn);
        }
    } else {
        float* out = (float*)outv;
        #pragma unroll
        for (int r = 0; r < 8; ++r) {
            const int orow = mt * 16 + half * 8 + r;
            out[(size_t)orow * D_ATT + n] = acc[r] + bn;
        }
    }
}

// ---------------------------------------------------------------------------
// Bilinear logits: out[b,q,k] = (qb[b,q,:] . kb[b,k,:]) / sqrt(128) + b_att
// One wave per 16x16 tile, K=128 -> 4 WMMAs. kb row-major acts as B^T directly.
// ---------------------------------------------------------------------------
__global__ __launch_bounds__(256) void bilinear_kernel(
    const _Float16* __restrict__ qb, const _Float16* __restrict__ kb,
    const float* __restrict__ b_att_p, float* __restrict__ out)
{
    const int lane = threadIdx.x & 31;
    const int wave = blockIdx.x * 8 + (threadIdx.x >> 5);
    const int nKT  = NK / 16;                  // 128
    const int nQT  = NQ / 16;                  // 16
    const int bIdx = wave / (nQT * nKT);
    const int rem  = wave % (nQT * nKT);
    const int qt   = rem / nKT;
    const int kt   = rem % nKT;
    const int half = lane >> 4;
    const int l15  = lane & 15;

    const _Float16* qrow = qb + ((size_t)bIdx * NQ + qt * 16 + l15) * D_ATT;
    const _Float16* krow = kb + ((size_t)bIdx * NK + kt * 16 + l15) * D_ATT;

    v8f acc = {};
    #pragma unroll
    for (int kk = 0; kk < D_ATT; kk += 32) {
        v8h a0 = *(const v8h*)(qrow + kk + half * 8);
        v8h a1 = *(const v8h*)(qrow + kk + 16 + half * 8);
        v8h b0 = *(const v8h*)(krow + kk + half * 8);
        v8h b1 = *(const v8h*)(krow + kk + 16 + half * 8);
        v16h a, b;
        #pragma unroll
        for (int j = 0; j < 8; ++j) {
            a[j] = a0[j]; a[j + 8] = a1[j];
            b[j] = b0[j]; b[j + 8] = b1[j];
        }
        acc = __builtin_amdgcn_wmma_f32_16x16x32_f16(
            false, a, false, b, (short)0, acc, false, false);
    }

    const float scale = 0.088388347648318447f;   // 1/sqrt(128)
    const float batt  = *b_att_p;
    float* obase = out + ((size_t)bIdx * NQ + qt * 16 + half * 8) * NK + kt * 16 + l15;
    #pragma unroll
    for (int r = 0; r < 8; ++r)
        obase[(size_t)r * NK] = acc[r] * scale + batt;
}

// ---------------------------------------------------------------------------
// Additive (Bahdanau) path, accumulated into out:
//   out[b,q,k] += sum_d v[d] * tanh(ql[b,q,d] + kl[b,k,d])
// Block = 16 q-rows x 32 k-rows tile, staged through LDS; each thread owns
// two (q,k) pairs sharing the same kl row. Transcendental-bound inner loop.
// ---------------------------------------------------------------------------
__global__ __launch_bounds__(256) void addlin_kernel(
    const float* __restrict__ ql, const float* __restrict__ kl,
    const float* __restrict__ v_att, float* __restrict__ out)
{
    __shared__ float qs[16][D_ATT + 4];
    __shared__ float ks[32][D_ATT + 4];
    __shared__ float vv[D_ATT];

    const int tid = threadIdx.x;
    const int nKT = NK / 32;                   // 64
    const int nQT = NQ / 16;                   // 16
    const int bIdx = blockIdx.x / (nQT * nKT);
    const int rem  = blockIdx.x % (nQT * nKT);
    const int qt   = rem / nKT;
    const int kt   = rem % nKT;

    const float* qg = ql + ((size_t)bIdx * NQ + qt * 16) * D_ATT;
    const float* kg = kl + ((size_t)bIdx * NK + kt * 32) * D_ATT;

    {   // stage ql tile: 16x128, 8 floats/thread
        const int row = tid >> 4;
        const int col = (tid & 15) * 8;
        *(float4*)&qs[row][col]     = *(const float4*)(qg + (size_t)row * D_ATT + col);
        *(float4*)&qs[row][col + 4] = *(const float4*)(qg + (size_t)row * D_ATT + col + 4);
    }
    {   // stage kl tile: 32x128, 16 floats/thread
        const int row = tid >> 3;
        const int col = (tid & 7) * 16;
        #pragma unroll
        for (int c = 0; c < 16; c += 4)
            *(float4*)&ks[row][col + c] = *(const float4*)(kg + (size_t)row * D_ATT + col + c);
    }
    if (tid < D_ATT) vv[tid] = v_att[tid];
    __syncthreads();

    const int ki = tid & 31;
    const int qi = tid >> 5;                   // 0..7 ; also handles qi+8
    float acc0 = 0.f, acc1 = 0.f;

    #pragma unroll 4
    for (int d = 0; d < D_ATT; d += 4) {
        float4 kv = *(const float4*)&ks[ki][d];
        float4 q0 = *(const float4*)&qs[qi][d];
        float4 q1 = *(const float4*)&qs[qi + 8][d];
        float4 v4 = *(const float4*)&vv[d];
        acc0 += v4.x * fast_tanh(q0.x + kv.x);
        acc1 += v4.x * fast_tanh(q1.x + kv.x);
        acc0 += v4.y * fast_tanh(q0.y + kv.y);
        acc1 += v4.y * fast_tanh(q1.y + kv.y);
        acc0 += v4.z * fast_tanh(q0.z + kv.z);
        acc1 += v4.z * fast_tanh(q1.z + kv.z);
        acc0 += v4.w * fast_tanh(q0.w + kv.w);
        acc1 += v4.w * fast_tanh(q1.w + kv.w);
    }

    const size_t base = ((size_t)bIdx * NQ + qt * 16) * NK + (size_t)kt * 32 + ki;
    out[base + (size_t)qi * NK]       += acc0;
    out[base + (size_t)(qi + 8) * NK] += acc1;
}

// ---------------------------------------------------------------------------
extern "C" void kernel_launch(void* const* d_in, const int* in_sizes, int n_in,
                              void* d_out, int out_size, void* d_ws, size_t ws_size,
                              hipStream_t stream) {
    (void)in_sizes; (void)n_in; (void)out_size; (void)ws_size;

    const float* query = (const float*)d_in[0];
    const float* key   = (const float*)d_in[1];
    const float* W_bq  = (const float*)d_in[2];
    const float* b_bq  = (const float*)d_in[3];
    const float* W_bk  = (const float*)d_in[4];
    const float* b_bk  = (const float*)d_in[5];
    const float* W_lq  = (const float*)d_in[6];
    const float* b_lq  = (const float*)d_in[7];
    const float* W_lk  = (const float*)d_in[8];
    const float* b_lk  = (const float*)d_in[9];
    const float* v_att = (const float*)d_in[10];
    const float* b_att = (const float*)d_in[11];
    float* out = (float*)d_out;

    // Workspace carve-up (~3.9 MB)
    const size_t WPSZ = (size_t)D_IN * D_ATT;            // 65536 halfs per packed W
    char* ws = (char*)d_ws;
    float*    ql   = (float*)ws;     ws += (size_t)BSZ * NQ * D_ATT * sizeof(float);
    float*    kl   = (float*)ws;     ws += (size_t)BSZ * NK * D_ATT * sizeof(float);
    _Float16* qbh  = (_Float16*)ws;  ws += (size_t)BSZ * NQ * D_ATT * sizeof(_Float16);
    _Float16* kbh  = (_Float16*)ws;  ws += (size_t)BSZ * NK * D_ATT * sizeof(_Float16);
    _Float16* WpLQ = (_Float16*)ws;  ws += WPSZ * sizeof(_Float16);
    _Float16* WpLK = (_Float16*)ws;  ws += WPSZ * sizeof(_Float16);
    _Float16* WpBQ = (_Float16*)ws;  ws += WPSZ * sizeof(_Float16);
    _Float16* WpBK = (_Float16*)ws;

    // Pack weights into WMMA B-fragment layout (f16)
    packw_kernel<<<256, 256, 0, stream>>>(W_lq, WpLQ);
    packw_kernel<<<256, 256, 0, stream>>>(W_lk, WpLK);
    packw_kernel<<<256, 256, 0, stream>>>(W_bq, WpBQ);
    packw_kernel<<<256, 256, 0, stream>>>(W_bk, WpBK);

    // Projections (WMMA): grid = M/16 blocks, 8 waves/block = (M/16)*8 tiles
    proj_kernel<0><<<(BSZ * NQ) / 16, 256, 0, stream>>>(query, WpLQ, b_lq, ql);
    proj_kernel<0><<<(BSZ * NK) / 16, 256, 0, stream>>>(key,   WpLK, b_lk, kl);
    proj_kernel<1><<<(BSZ * NQ) / 16, 256, 0, stream>>>(query, WpBQ, b_bq, qbh);
    proj_kernel<1><<<(BSZ * NK) / 16, 256, 0, stream>>>(key,   WpBK, b_bk, kbh);

    // Bilinear logits (WMMA): 2*16*128 = 4096 tiles, 8 waves/block -> 512 blocks
    bilinear_kernel<<<(BSZ * (NQ / 16) * (NK / 16)) / 8, 256, 0, stream>>>(
        qbh, kbh, b_att, out);

    // Additive tanh path: 2*16*64 = 2048 blocks
    addlin_kernel<<<BSZ * (NQ / 16) * (NK / 32), 256, 0, stream>>>(
        ql, kl, v_att, out);
}
